// Vmamba1DBlock_90237262889283
// MI455X (gfx1250) — matrix operand order repaired and checked by reading, not compiled, not run on hardware
//
#include <hip/hip_runtime.h>
#include <stdint.h>

// ---------------------------------------------------------------------------
// VMamba 1D block on MI455X (gfx1250), wave32 + WMMA + async-to-LDS.
//   All 6 projection GEMMs: v_wmma_f32_16x16x32_bf16, fp32 accumulate,
//   register double-buffered K-loop (K % 64 == 0 for every GEMM here).
//   Selective scan: fp32, h[16] in registers; per-step B/C vectors prefetched
//   with global_load_async_to_lds_b32 (ASYNCcnt) one step ahead; dt/u streams
//   register-pipelined; one barrier per timestep.
// ---------------------------------------------------------------------------

#define D_MODEL   768
#define D_STATE   16
#define D_INNER   1536
#define DT_RANK   48
#define DT_PAD    64          // K=48 padded to 64 (multiple of 64) with zeros
#define KDIR      2
#define B_SZ      2
#define L_SEQ     2048
#define M_ROWS    (B_SZ * L_SEQ)            // 4096
#define XPROJ_N   (DT_RANK + 2 * D_STATE)   // 80

typedef __bf16 bf16;
typedef __attribute__((ext_vector_type(8)))  __bf16 v8bf;
typedef __attribute__((ext_vector_type(16))) __bf16 v16bf;
typedef __attribute__((ext_vector_type(8)))  float  v8f;

__device__ __forceinline__ unsigned short f32_to_bf16_bits(float f) {
  unsigned int x = __float_as_uint(f);
  unsigned int r = x + 0x7FFFu + ((x >> 16) & 1u);   // round-to-nearest-even
  return (unsigned short)(r >> 16);
}
__device__ __forceinline__ float bf16_bits_to_f32(unsigned short u) {
  return __uint_as_float(((unsigned int)u) << 16);
}

// async DMA: 4 bytes per active lane, global -> LDS (ASYNCcnt-tracked)
__device__ __forceinline__ void async_load_f32_to_lds(const float* gptr, void* lptr) {
  unsigned lds = (unsigned)(uintptr_t)lptr;          // low 32b = LDS offset
  uint64_t ga  = (uint64_t)(uintptr_t)gptr;
  asm volatile("global_load_async_to_lds_b32 %0, %1, off"
               :: "v"(lds), "v"(ga) : "memory");
}
__device__ __forceinline__ void wait_asynccnt0() {
  asm volatile("s_wait_asynccnt 0x0" ::: "memory");
}

// ------------------------- small elementwise kernels -----------------------

__global__ __launch_bounds__(256) void cvt_f32_bf16_kernel(
    const float* __restrict__ src, unsigned short* __restrict__ dst, int n) {
  int idx = blockIdx.x * 256 + threadIdx.x;
  if (idx < n) dst[idx] = f32_to_bf16_bits(src[idx]);
}

// dt_proj_w [KDIR, D_INNER, 48] -> bf16 [KDIR, D_INNER, 64] zero-padded
__global__ __launch_bounds__(256) void pad_dtw_kernel(
    const float* __restrict__ src, unsigned short* __restrict__ dst) {
  int idx = blockIdx.x * 256 + threadIdx.x;
  if (idx >= KDIR * D_INNER * DT_PAD) return;
  int c  = idx & (DT_PAD - 1);
  int kd = idx >> 6;
  float v = (c < DT_RANK) ? src[(size_t)kd * DT_RANK + c] : 0.f;
  dst[idx] = f32_to_bf16_bits(v);
}

// x_dbl [M_ROWS, 80] -> dt_in bf16 [M_ROWS, 64] (cols 0..47, rest zero)
__global__ __launch_bounds__(256) void dtin_kernel(
    const float* __restrict__ x_dbl, unsigned short* __restrict__ dtin) {
  int idx = blockIdx.x * 256 + threadIdx.x;
  if (idx >= M_ROWS * DT_PAD) return;
  int c   = idx & (DT_PAD - 1);
  int row = idx >> 6;
  float v = (c < DT_RANK) ? x_dbl[(size_t)row * XPROJ_N + c] : 0.f;
  dtin[idx] = f32_to_bf16_bits(v);
}

// depthwise conv(k=3, pad=1) over L + bias + silu -> u (bf16, fwd & reversed)
__global__ __launch_bounds__(256) void conv_silu_kernel(
    const float* __restrict__ xz, const float* __restrict__ conv_w,
    const float* __restrict__ conv_b,
    unsigned short* __restrict__ u_bf, unsigned short* __restrict__ u_bf_rev) {
  int idx = blockIdx.x * 256 + threadIdx.x;
  if (idx >= M_ROWS * D_INNER) return;
  int d   = idx % D_INNER;
  int row = idx / D_INNER;
  int l   = row % L_SEQ;
  int b   = row / L_SEQ;
  float w0 = conv_w[d * 3 + 0], w1 = conv_w[d * 3 + 1], w2 = conv_w[d * 3 + 2];
  float acc = conv_b[d];
  const float* xc = xz + (size_t)row * (2 * D_INNER) + d;   // x_conv half of xz
  if (l > 0)         acc = fmaf(w0, xc[-(ptrdiff_t)(2 * D_INNER)], acc);
  acc = fmaf(w1, xc[0], acc);
  if (l < L_SEQ - 1) acc = fmaf(w2, xc[2 * D_INNER], acc);
  float s = acc / (1.f + __expf(-acc));                      // silu
  unsigned short v = f32_to_bf16_bits(s);
  u_bf[idx] = v;
  u_bf_rev[(size_t)(b * L_SEQ + (L_SEQ - 1 - l)) * D_INNER + d] = v;
}

// y_gated = y_sum * z  -> bf16 for the out_proj GEMM
__global__ __launch_bounds__(256) void gate_kernel(
    const float* __restrict__ y_sum, const float* __restrict__ xz,
    unsigned short* __restrict__ yg) {
  int idx = blockIdx.x * 256 + threadIdx.x;
  if (idx >= M_ROWS * D_INNER) return;
  int d   = idx % D_INNER;
  int row = idx / D_INNER;
  float z = xz[(size_t)row * (2 * D_INNER) + D_INNER + d];
  yg[idx] = f32_to_bf16_bits(y_sum[idx] * z);
}

// ------------------------------ WMMA GEMM ----------------------------------
// C[M,N] (f32) = A[M,K] (bf16, row-major) x W[N,K]^T (bf16, row-major).
// Wave tile 32x64 (2x4 of 16x16), block 8 waves -> 64x256 tile.
// A frag per lane: two contiguous 16B chunks at K = k0+h*8 and k0+16+h*8.
// B frag per lane: two contiguous 16B chunks of W[n] at K = k0+h*8 and k0+16+h*8.
// Out-of-range B rows are clamped (loads always issue, branch-free inner
// loop); the corresponding accumulators are garbage but never stored.
// K-loop is register double-buffered; requires Kd % 64 == 0 (768/1536/64).

__device__ __forceinline__ void load_frags(
    const bf16* __restrict__ a_base, const bf16* const* __restrict__ b_base,
    int Kd, int k0, v16bf a[2], v16bf b[4]) {
#pragma unroll
  for (int i = 0; i < 2; ++i) {
    const bf16* p = a_base + (size_t)i * 16 * Kd + k0;
    v8bf lo = *(const v8bf*)(p);        // K = k0 + h*8      .. +7
    v8bf hi = *(const v8bf*)(p + 16);   // K = k0 + 16 + h*8 .. +7
#pragma unroll
    for (int e = 0; e < 8; ++e) { a[i][e] = lo[e]; a[i][e + 8] = hi[e]; }
  }
#pragma unroll
  for (int j = 0; j < 4; ++j) {
    v8bf lo = *(const v8bf*)(b_base[j] + k0);        // K = k0 + h*8      .. +7
    v8bf hi = *(const v8bf*)(b_base[j] + k0 + 16);   // K = k0 + 16 + h*8 .. +7
#pragma unroll
    for (int e = 0; e < 8; ++e) { b[j][e] = lo[e]; b[j][e + 8] = hi[e]; }
  }
}

__device__ __forceinline__ void wmma8(v8f acc[2][4], const v16bf a[2],
                                      const v16bf b[4]) {
#pragma unroll
  for (int i = 0; i < 2; ++i)
#pragma unroll
    for (int j = 0; j < 4; ++j)
      acc[i][j] = __builtin_amdgcn_wmma_f32_16x16x32_bf16(
          false, a[i], false, b[j], (short)0, acc[i][j], false, false);
}

__global__ __launch_bounds__(256) void gemm_bf16_wmma(
    const unsigned short* __restrict__ Au, const unsigned short* __restrict__ Wu,
    float* __restrict__ C, int M, int N, int Kd) {
  const bf16* A = reinterpret_cast<const bf16*>(Au);
  const bf16* W = reinterpret_cast<const bf16*>(Wu);
  const int lane = threadIdx.x & 31;
  const int wave = threadIdx.x >> 5;
  const int h    = lane >> 4;       // lane half selects K sub-range
  const int ln   = lane & 15;       // A: row-in-tile, B/C: col-in-tile
  const int wm   = wave >> 2;       // 0..1
  const int wn   = wave & 3;        // 0..3
  const int m0 = blockIdx.y * 64 + wm * 32;
  const int n0 = blockIdx.x * 256 + wn * 64;

  v8f acc[2][4];
#pragma unroll
  for (int i = 0; i < 2; ++i)
#pragma unroll
    for (int j = 0; j < 4; ++j)
#pragma unroll
      for (int e = 0; e < 8; ++e) acc[i][j][e] = 0.f;

  bool actn[4];
#pragma unroll
  for (int j = 0; j < 4; ++j) actn[j] = (n0 + j * 16) < N;  // N mult of 16

  const bf16* a_base = A + (size_t)(m0 + ln) * Kd + h * 8;
  const bf16* b_base[4];
#pragma unroll
  for (int j = 0; j < 4; ++j) {
    int nc = n0 + j * 16 + ln;
    if (nc >= N) nc = N - 1;        // clamp: branch-free loads, gated stores
    b_base[j] = W + (size_t)nc * Kd + h * 8;
  }

  v16bf a0[2], b0[4], a1[2], b1[4];
  load_frags(a_base, b_base, Kd, 0, a0, b0);

  for (int k0 = 0; k0 < Kd; k0 += 64) {
    load_frags(a_base, b_base, Kd, k0 + 32, a1, b1);   // stage 1 in flight
    if (k0 + 64 < Kd) {
      __builtin_prefetch(a_base + (k0 + 64), 0, 3);    // global_prefetch_b8
      __builtin_prefetch(b_base[0] + (k0 + 64), 0, 3);
    }
    wmma8(acc, a0, b0);                                // consume stage 0
    if (k0 + 64 < Kd)
      load_frags(a_base, b_base, Kd, k0 + 64, a0, b0); // refill stage 0
    wmma8(acc, a1, b1);                                // consume stage 1
  }

#pragma unroll
  for (int i = 0; i < 2; ++i)
#pragma unroll
    for (int j = 0; j < 4; ++j) {
      if (!actn[j]) continue;
      int nc = n0 + j * 16 + ln;
      size_t base = (size_t)(m0 + i * 16 + 8 * h) * N + nc;   // m = r + 8h
#pragma unroll
      for (int r = 0; r < 8; ++r) C[base + (size_t)r * N] = acc[i][j][r];
    }
}

// --------------------------- selective scan --------------------------------
// One thread per (b, d) channel, h[16] in registers, sequential over L.
// B_t/C_t (32 floats shared by all d) are DMA'd global->LDS one step ahead
// by wave 0 via global_load_async_to_lds_b32; dt/u streams are register-
// pipelined one step ahead. One __syncthreads per timestep.
__global__ __launch_bounds__(256) void scan_kernel(
    const float* __restrict__ dt_raw, const float* __restrict__ x_dbl,
    const unsigned short* __restrict__ u_dir,   // direction-ordered u (bf16)
    const float* __restrict__ dt_b, const float* __restrict__ A_logs,
    const float* __restrict__ Ds, float* __restrict__ y_sum, int dir) {
  const int chunks = D_INNER / 256;
  int b  = blockIdx.x / chunks;
  int d  = (blockIdx.x % chunks) * 256 + threadIdx.x;
  int gd = dir * D_INNER + d;

  float A[D_STATE];
#pragma unroll
  for (int n = 0; n < D_STATE; ++n)
    A[n] = -__expf(A_logs[(size_t)gd * D_STATE + n]);
  float Dk   = Ds[gd];
  float bias = dt_b[gd];
  float hst[D_STATE];
#pragma unroll
  for (int n = 0; n < D_STATE; ++n) hst[n] = 0.f;

  __shared__ float sBC[2][2 * D_STATE];       // double-buffered B|C vectors
  const size_t rowbase = (size_t)b * L_SEQ;

  // prologue: DMA B/C(t=0) into buffer 0; preload dt/u(t=0) into registers
  if (threadIdx.x < 2 * D_STATE)
    async_load_f32_to_lds(
        x_dbl + rowbase * XPROJ_N + DT_RANK + threadIdx.x, &sBC[0][threadIdx.x]);
  float dt_cur = dt_raw[rowbase * D_INNER + d];
  float u_cur  = bf16_bits_to_f32(u_dir[rowbase * D_INNER + d]);
  float dt_nxt = 0.f, u_nxt = 0.f;

  for (int t = 0; t < L_SEQ; ++t) {
    const int p = t & 1;
    wait_asynccnt0();            // per-wave counter: no-op for waves 1..7
    __syncthreads();             // buf[p] visible; buf[1-p] readers done

    if (t + 1 < L_SEQ) {         // prefetch step t+1
      size_t nrow = rowbase + t + 1;
      if (threadIdx.x < 2 * D_STATE)
        async_load_f32_to_lds(
            x_dbl + nrow * XPROJ_N + DT_RANK + threadIdx.x,
            &sBC[1 - p][threadIdx.x]);
      dt_nxt = dt_raw[nrow * D_INNER + d];
      u_nxt  = bf16_bits_to_f32(u_dir[nrow * D_INNER + d]);
    }

    float dtv   = dt_cur + bias;
    float delta = (dtv > 20.f) ? dtv : log1pf(__expf(dtv));   // softplus
    float du    = delta * u_cur;
    float yt    = 0.f;
#pragma unroll
    for (int n = 0; n < D_STATE; ++n) {
      float dA = __expf(delta * A[n]);
      hst[n] = fmaf(hst[n], dA, du * sBC[p][n]);
      yt = fmaf(hst[n], sBC[p][D_STATE + n], yt);
    }
    float yo   = yt + u_cur * Dk;
    int   orow = dir ? (int)(rowbase + (L_SEQ - 1 - t)) : (int)(rowbase + t);
    if (dir) y_sum[(size_t)orow * D_INNER + d] += yo;   // flip-add dir 1
    else     y_sum[(size_t)orow * D_INNER + d]  = yo;

    dt_cur = dt_nxt;
    u_cur  = u_nxt;
  }
}

// ------------------------------- launcher ----------------------------------

extern "C" void kernel_launch(void* const* d_in, const int* in_sizes, int n_in,
                              void* d_out, int out_size, void* d_ws, size_t ws_size,
                              hipStream_t stream) {
  (void)in_sizes; (void)n_in; (void)out_size; (void)ws_size;
  const float* x       = (const float*)d_in[0];
  const float* in_w    = (const float*)d_in[1];
  const float* conv_w  = (const float*)d_in[2];
  const float* conv_b  = (const float*)d_in[3];
  const float* xproj_w = (const float*)d_in[4];
  const float* dtw     = (const float*)d_in[5];
  const float* dtb     = (const float*)d_in[6];
  const float* A_logs  = (const float*)d_in[7];
  const float* Ds      = (const float*)d_in[8];
  const float* out_w   = (const float*)d_in[9];
  float* out = (float*)d_out;

  char* ws = (char*)d_ws;
  size_t off = 0;
  auto alloc = [&](size_t bytes) -> void* {
    void* p = ws + off;
    off = (off + bytes + 255) & ~(size_t)255;
    return p;
  };
  unsigned short* xbf    = (unsigned short*)alloc((size_t)M_ROWS * D_MODEL * 2);
  unsigned short* w1bf   = (unsigned short*)alloc((size_t)2 * D_INNER * D_MODEL * 2);
  unsigned short* w3bf   = (unsigned short*)alloc((size_t)D_MODEL * D_INNER * 2);
  unsigned short* w2bf   = (unsigned short*)alloc((size_t)KDIR * XPROJ_N * D_INNER * 2);
  unsigned short* wdtbf  = (unsigned short*)alloc((size_t)KDIR * D_INNER * DT_PAD * 2);
  float*          xz     = (float*)alloc((size_t)M_ROWS * 2 * D_INNER * 4);
  unsigned short* u_bf   = (unsigned short*)alloc((size_t)M_ROWS * D_INNER * 2);
  unsigned short* u_bfr  = (unsigned short*)alloc((size_t)M_ROWS * D_INNER * 2);
  float*          x_dbl  = (float*)alloc((size_t)M_ROWS * XPROJ_N * 4);
  unsigned short* dtin   = (unsigned short*)alloc((size_t)M_ROWS * DT_PAD * 2);
  float*          dt_raw = (float*)alloc((size_t)M_ROWS * D_INNER * 4);
  float*          y_sum  = (float*)alloc((size_t)M_ROWS * D_INNER * 4);
  unsigned short* yg_bf  = (unsigned short*)alloc((size_t)M_ROWS * D_INNER * 2);

  auto cvt = [&](const float* s, unsigned short* d, int n) {
    cvt_f32_bf16_kernel<<<dim3((n + 255) / 256), dim3(256), 0, stream>>>(s, d, n);
  };
  auto gemm = [&](const unsigned short* A, const unsigned short* W, float* C,
                  int M, int N, int K) {
    dim3 grid((N + 255) / 256, M / 64);
    gemm_bf16_wmma<<<grid, dim3(256), 0, stream>>>(A, W, C, M, N, K);
  };

  // --- precision staging: f32 -> bf16 operands ---
  cvt(x,       xbf,  M_ROWS * D_MODEL);
  cvt(in_w,    w1bf, 2 * D_INNER * D_MODEL);
  cvt(out_w,   w3bf, D_MODEL * D_INNER);
  cvt(xproj_w, w2bf, KDIR * XPROJ_N * D_INNER);
  pad_dtw_kernel<<<dim3((KDIR * D_INNER * DT_PAD + 255) / 256), dim3(256), 0, stream>>>(dtw, wdtbf);

  // --- in_proj: xz[4096,3072] = x[4096,768] @ in_w^T ---
  gemm(xbf, w1bf, xz, M_ROWS, 2 * D_INNER, D_MODEL);

  // --- depthwise conv + silu -> u (fwd + reversed, bf16) ---
  {
    int n = M_ROWS * D_INNER;
    conv_silu_kernel<<<dim3((n + 255) / 256), dim3(256), 0, stream>>>(
        xz, conv_w, conv_b, u_bf, u_bfr);
  }

  // --- per-direction: x_proj GEMM -> dt GEMM -> selective scan ---
  for (int dir = 0; dir < KDIR; ++dir) {
    const unsigned short* u_dir = dir ? u_bfr : u_bf;
    gemm(u_dir, w2bf + (size_t)dir * XPROJ_N * D_INNER, x_dbl,
         M_ROWS, XPROJ_N, D_INNER);
    dtin_kernel<<<dim3((M_ROWS * DT_PAD + 255) / 256), dim3(256), 0, stream>>>(x_dbl, dtin);
    gemm(dtin, wdtbf + (size_t)dir * D_INNER * DT_PAD, dt_raw,
         M_ROWS, D_INNER, DT_PAD);
    scan_kernel<<<dim3(B_SZ * (D_INNER / 256)), dim3(256), 0, stream>>>(
        dt_raw, x_dbl, u_dir, dtb, A_logs, Ds, y_sum, dir);
  }

  // --- gating + out_proj ---
  {
    int n = M_ROWS * D_INNER;
    gate_kernel<<<dim3((n + 255) / 256), dim3(256), 0, stream>>>(y_sum, xz, yg_bf);
  }
  gemm(yg_bf, w3bf, out, M_ROWS, D_MODEL, D_INNER);
}
